// EfficientMultiHeadAttention_45260365365406
// MI455X (gfx1250) — matrix operand order repaired and checked by reading, not compiled, not run on hardware
//
#include <hip/hip_runtime.h>
#include <math.h>

// ---------------------------------------------------------------------------
// EfficientMultiHeadAttention (Linformer) for MI455X / gfx1250, wave32 + WMMA.
// B=4, C=64, H=W=64 -> T=4096, HEADS=8, dh=8, rank k=1024.
// Dead conv+LN branch of the reference is skipped (result never used).
// GEMMs: v_wmma_f32_16x16x32_bf16 (bf16 in, f32 accumulate).
// Staging: global->LDS async copies (gfx1250 ASYNCcnt path).
// ---------------------------------------------------------------------------

typedef __bf16 bf16;
typedef __attribute__((ext_vector_type(16))) __bf16 v16bf;
typedef __attribute__((ext_vector_type(8)))  float  v8f;
typedef __attribute__((ext_vector_type(4)))  float  v4f;
typedef __attribute__((ext_vector_type(4)))  int    v4i;

#define WMMA_BF16(A, B, C) \
  __builtin_amdgcn_wmma_f32_16x16x32_bf16(false, (A), false, (B), (short)0, (C), false, false)

static __device__ inline v8f v8zero() {
  v8f z = {0.f, 0.f, 0.f, 0.f, 0.f, 0.f, 0.f, 0.f};
  return z;
}

// ---------------------------------------------------------------------------
// Async global->LDS staging. Builtin signature (from hipcc diagnostic):
//   void __builtin_amdgcn_global_load_async_to_lds_b128(v4i AS1* src,
//                                                       v4i AS3* dst,
//                                                       imm int offset,
//                                                       imm int cpol)
// ---------------------------------------------------------------------------
#if defined(__has_builtin)
#if __has_builtin(__builtin_amdgcn_global_load_async_to_lds_b128) && \
    __has_builtin(__builtin_amdgcn_s_wait_asynccnt)
#define USE_ASYNC 1
#endif
#endif
#ifndef USE_ASYNC
#define USE_ASYNC 0
#endif

typedef __attribute__((address_space(3))) v4i lds_v4i;
typedef __attribute__((address_space(1))) v4i glb_v4i;

static __device__ inline lds_v4i* lds_cast(void* p) {
  // Generic LDS address: aperture lives in the high 32 bits; addr[31:0] is the
  // LDS byte offset (ISA 10.2), so truncation yields a valid AS(3) pointer.
  return (lds_v4i*)(unsigned int)(unsigned long long)p;
}
static __device__ inline glb_v4i* glb_cast(const float* p) {
  return (glb_v4i*)(unsigned long long)p;
}

// Copy 16 bytes global -> LDS (async when available).
static __device__ inline void g2l_16B(void* lds_dst, const float* g_src) {
#if USE_ASYNC
  __builtin_amdgcn_global_load_async_to_lds_b128(glb_cast(g_src), lds_cast(lds_dst), 0, 0);
#else
  *(v4f*)lds_dst = *(const v4f*)g_src;
#endif
}
static __device__ inline void stage_wait() {
#if USE_ASYNC
  __builtin_amdgcn_s_wait_asynccnt(0);
#endif
}

// WMMA fragment index maps (wave32, hand-packed per CDNA5 ISA):
//  A (16x32 bf16): lane L -> row M=L&15; elem e -> K = e + (e>=8?8:0) + 8*(L>>4)
//  B (32x16 bf16): lane L -> col N=L&15; elem e -> K = e + (e>=8?8:0) + 8*(L>>4)
//  D (16x16 f32):  elem g -> row M = g + 8*(L>>4), col N = L&15

// ---------------------------------------------------------------------------
// Kernel 1: qkv = xf @ Wqkv. One wave per 16-row m-tile, all 12 n-tiles
// (x read exactly once). Wqkv staged once per block in LDS as bf16.
// ---------------------------------------------------------------------------
__global__ void qkv_wmma(const float* __restrict__ x, const float* __restrict__ Wqkv,
                         float* __restrict__ Qb, float* __restrict__ Kb,
                         float* __restrict__ Vb) {
  __shared__ bf16 ldsW[64 * 192];                  // 24 KB
  const int tid = threadIdx.x;
  for (int i = tid; i < 64 * 192; i += 256) ldsW[i] = (bf16)Wqkv[i];
  __syncthreads();

  const int lane = tid & 31;
  const int half = lane >> 4;
  const int l    = lane & 15;
  const int mt   = blockIdx.x * 8 + (tid >> 5);    // 1024 m-tiles over (b, t)
  const int b    = mt >> 8;
  const int t0   = (mt & 255) * 16;

  v16bf Av[2];
#pragma unroll
  for (int kk2 = 0; kk2 < 2; ++kk2)
#pragma unroll
    for (int e = 0; e < 16; ++e) {
      const int ka = kk2 * 32 + e + ((e >= 8) ? 8 : 0) + 8 * half;   // channel c
      Av[kk2][e] = (bf16)x[(b * 64 + ka) * 4096 + t0 + l];           // xf[b][t][c]=x[b][c][t]
    }

  v8f acc[12];
#pragma unroll
  for (int nt = 0; nt < 12; ++nt) acc[nt] = v8zero();

#pragma unroll
  for (int nt = 0; nt < 12; ++nt)
#pragma unroll
    for (int kk2 = 0; kk2 < 2; ++kk2) {
      v16bf Bv;
#pragma unroll
      for (int e = 0; e < 16; ++e) {
        const int kb = kk2 * 32 + e + ((e >= 8) ? 8 : 0) + 8 * half;
        Bv[e] = ldsW[kb * 192 + nt * 16 + l];
      }
      acc[nt] = WMMA_BF16(Av[kk2], Bv, acc[nt]);
    }

  // Column j in [0,192): j = d*24 + which*8 + h  (reference reshape (dh,3,HEADS))
#pragma unroll
  for (int nt = 0; nt < 12; ++nt) {
    const int j     = nt * 16 + l;
    const int d     = j / 24;
    const int rem   = j % 24;
    const int which = rem >> 3;
    const int hh    = rem & 7;
    const int cp    = hh * 8 + d;                  // head-major channel (h d)
    float* dstp = (which == 0) ? Qb : (which == 1) ? Kb : Vb;
#pragma unroll
    for (int g = 0; g < 8; ++g)
      dstp[(b * 4096 + t0 + g + 8 * half) * 64 + cp] = acc[nt][g];
  }
}

// ---------------------------------------------------------------------------
// Kernel 2: {Kp,Vp}[b][r][c'] = sum_j E[j][r]*{K,V}[b][j][c'].  K = 4096.
// Block = 8 waves, 128 (r) x 64 (c') tile. Double-buffered LDS staging of the
// E^T slice (32x128) and K/V slice (32x64) via async global->LDS copies.
// ---------------------------------------------------------------------------
__global__ void linformer_proj_wmma(const float* __restrict__ E,
                                    const float* __restrict__ Kb,
                                    const float* __restrict__ Vb,
                                    float* __restrict__ Kp, float* __restrict__ Vp) {
  __shared__ float ldsA[2][32 * 128];              // 32 KB: E[j][r] slice
  __shared__ float ldsB[2][32 * 64];               // 16 KB: K/V slice

  const int mseg = blockIdx.x & 7;                 // 8 segments of 128 rows
  const int mat  = (blockIdx.x >> 3) & 1;          // 0 = K, 1 = V
  const int b    = blockIdx.x >> 4;
  const int r0   = mseg * 128;

  const float* __restrict__ src = mat ? Vb : Kb;
  float* __restrict__ dst       = mat ? Vp : Kp;

  const int tid  = threadIdx.x;
  const int w    = tid >> 5;
  const int lane = tid & 31;
  const int half = lane >> 4;
  const int l    = lane & 15;

  // Stage k-chunk [kk, kk+32) into buffer `buf`.
  auto stage = [&](int kk, int buf) {
    // A: 32 rows x 128 floats (row stride 1024 in E), 1024 float4s.
#pragma unroll
    for (int it = 0; it < 4; ++it) {
      const int f4 = tid + it * 256;
      const int kr = f4 >> 5;
      const int rc = (f4 & 31) << 2;
      g2l_16B(&ldsA[buf][kr * 128 + rc], &E[(kk + kr) * 1024 + r0 + rc]);
    }
    // B: 32 rows x 64 floats, fully contiguous (2048 floats), 512 float4s.
#pragma unroll
    for (int it = 0; it < 2; ++it) {
      const int f4 = tid + it * 256;
      g2l_16B(&ldsB[buf][f4 * 4], &src[(b * 4096 + kk) * 64 + f4 * 4]);
    }
  };

  v8f acc[4];
#pragma unroll
  for (int nt = 0; nt < 4; ++nt) acc[nt] = v8zero();

  stage(0, 0);
  for (int step = 0; step < 128; ++step) {
    const int cur = step & 1;
    stage_wait();                                  // my wave's async copies done
    __syncthreads();                               // all waves' copies visible
    if (step + 1 < 128) stage((step + 1) * 32, cur ^ 1);

    v16bf Av;
#pragma unroll
    for (int e = 0; e < 16; ++e) {
      const int ka = e + ((e >= 8) ? 8 : 0) + 8 * half;
      Av[e] = (bf16)ldsA[cur][ka * 128 + w * 16 + l];
    }
#pragma unroll
    for (int nt = 0; nt < 4; ++nt) {
      v16bf Bv;
#pragma unroll
      for (int e = 0; e < 16; ++e) {
        const int kb = e + ((e >= 8) ? 8 : 0) + 8 * half;
        Bv[e] = (bf16)ldsB[cur][kb * 64 + nt * 16 + l];
      }
      acc[nt] = WMMA_BF16(Av, Bv, acc[nt]);
    }
    __syncthreads();                               // done reading `cur`
  }

#pragma unroll
  for (int nt = 0; nt < 4; ++nt)
#pragma unroll
    for (int g = 0; g < 8; ++g)
      dst[(b * 1024 + r0 + w * 16 + g + 8 * half) * 64 + nt * 16 + l] = acc[nt][g];
}

// ---------------------------------------------------------------------------
// Kernel 3: attention with online softmax (scores never materialized).
// One block per (b, h, 256-row chunk); Kp/Vp head slices async-staged to LDS.
// dh = 8 is too thin for WMMA; thread-per-row VALU + LDS is the right shape.
// ---------------------------------------------------------------------------
__global__ void attn_kernel(const float* __restrict__ Qb, const float* __restrict__ Kp,
                            const float* __restrict__ Vp, float* __restrict__ Ob) {
  __shared__ float kph[1024 * 8];
  __shared__ float vph[1024 * 8];

  const int chunk = blockIdx.x & 15;
  const int h     = (blockIdx.x >> 4) & 7;
  const int b     = blockIdx.x >> 7;

  // 2048 float4s per array; rows are 32B-aligned (h*8 floats into 64-float rows).
  for (int f4 = threadIdx.x; f4 < 2048; f4 += blockDim.x) {
    const int r  = f4 >> 1;
    const int d4 = (f4 & 1) << 2;
    g2l_16B(&kph[r * 8 + d4], &Kp[(b * 1024 + r) * 64 + h * 8 + d4]);
    g2l_16B(&vph[r * 8 + d4], &Vp[(b * 1024 + r) * 64 + h * 8 + d4]);
  }
  stage_wait();
  __syncthreads();

  const int t = chunk * 256 + threadIdx.x;
  float q[8];
#pragma unroll
  for (int d = 0; d < 8; ++d) q[d] = Qb[(b * 4096 + t) * 64 + h * 8 + d];

  const float scale = 0.35355339059327373f;        // 8^-0.5
  float m = -INFINITY, ssum = 0.f;
  float o[8];
#pragma unroll
  for (int d = 0; d < 8; ++d) o[d] = 0.f;

  for (int r = 0; r < 1024; ++r) {
    float s = 0.f;
#pragma unroll
    for (int d = 0; d < 8; ++d) s += q[d] * kph[r * 8 + d];
    s *= scale;
    const float mn   = fmaxf(m, s);
    const float corr = __expf(m - mn);
    const float p    = __expf(s - mn);
    ssum = ssum * corr + p;
#pragma unroll
    for (int d = 0; d < 8; ++d) o[d] = o[d] * corr + p * vph[r * 8 + d];
    m = mn;
  }

  const float inv = 1.f / ssum;
#pragma unroll
  for (int d = 0; d < 8; ++d)
    Ob[(b * 4096 + t) * 64 + h * 8 + d] = o[d] * inv;
}

// ---------------------------------------------------------------------------
// Kernel 4: y = attn_out @ W0, fused final (b,t,c) -> NCHW transpose on store.
// W0 staged in LDS as bf16. One wave per 16 (t) x 64 (c2) strip, K = 64.
// ---------------------------------------------------------------------------
__global__ void outproj_wmma(const float* __restrict__ Ob, const float* __restrict__ W0,
                             float* __restrict__ out) {
  __shared__ bf16 ldsW0[64 * 64];                  // 8 KB
  const int tid = threadIdx.x;
  for (int i = tid; i < 64 * 64; i += 256) ldsW0[i] = (bf16)W0[i];
  __syncthreads();

  const int lane = tid & 31;
  const int half = lane >> 4;
  const int l    = lane & 15;
  const int mt   = blockIdx.x * 8 + (tid >> 5);    // 1024 m-tiles over (b, t)
  const int b    = mt >> 8;
  const int t0   = (mt & 255) * 16;

  v8f acc[4];
#pragma unroll
  for (int nt = 0; nt < 4; ++nt) acc[nt] = v8zero();

#pragma unroll
  for (int kk = 0; kk < 64; kk += 32) {
    v16bf Av;
#pragma unroll
    for (int e = 0; e < 16; ++e) {
      const int ka = kk + e + ((e >= 8) ? 8 : 0) + 8 * half;       // channel c'
      Av[e] = (bf16)Ob[(b * 4096 + t0 + l) * 64 + ka];
    }
#pragma unroll
    for (int nt = 0; nt < 4; ++nt) {
      v16bf Bv;
#pragma unroll
      for (int e = 0; e < 16; ++e) {
        const int kb = kk + e + ((e >= 8) ? 8 : 0) + 8 * half;
        Bv[e] = ldsW0[kb * 64 + nt * 16 + l];
      }
      acc[nt] = WMMA_BF16(Av, Bv, acc[nt]);
    }
  }

#pragma unroll
  for (int nt = 0; nt < 4; ++nt)
#pragma unroll
    for (int g = 0; g < 8; ++g)
      out[(b * 64 + nt * 16 + l) * 4096 + t0 + g + 8 * half] = acc[nt][g];
}

// ---------------------------------------------------------------------------
// Launch. Inputs: x, conv_w, conv_b, ln_g, ln_b, Wqkv, W0, E (conv/LN unused).
// Workspace: Qb/Kb/Vb [4][4096][64], Kp/Vp [4][1024][64], Ob [4][4096][64]
//   -> 4,718,592 floats (~18 MB).
// ---------------------------------------------------------------------------
extern "C" void kernel_launch(void* const* d_in, const int* in_sizes, int n_in,
                              void* d_out, int out_size, void* d_ws, size_t ws_size,
                              hipStream_t stream) {
  const float* x    = (const float*)d_in[0];
  const float* Wqkv = (const float*)d_in[5];
  const float* W0   = (const float*)d_in[6];
  const float* E    = (const float*)d_in[7];
  float* out = (float*)d_out;

  float* ws = (float*)d_ws;
  float* Qb = ws;                        // 4*4096*64
  float* Kb = Qb + 4 * 4096 * 64;
  float* Vb = Kb + 4 * 4096 * 64;
  float* Kp = Vb + 4 * 4096 * 64;        // 4*1024*64
  float* Vp = Kp + 4 * 1024 * 64;
  float* Ob = Vp + 4 * 1024 * 64;

  // 1024 waves (one per m-tile), 8 per block
  qkv_wmma<<<128, 256, 0, stream>>>(x, Wqkv, Qb, Kb, Vb);
  // one block per (b, mat, 128-row segment)
  linformer_proj_wmma<<<64, 256, 0, stream>>>(E, Kb, Vb, Kp, Vp);
  // one block per (b, h, 256-row chunk)
  attn_kernel<<<512, 256, 0, stream>>>(Qb, Kp, Vp, Ob);
  // 1024 waves, 8 per block
  outproj_wmma<<<128, 256, 0, stream>>>(Ob, W0, out);
}
